// MixtureOfExperts_47596827574641
// MI455X (gfx1250) — compile-verified, matrix operand order checked by reading, not run on hardware
//
#include <hip/hip_runtime.h>
#include <hip/hip_bf16.h>
#include <stdint.h>

#define DDIM 1024
#define FDIM 4096
#define NEXP 8
#define NTOK 4096
#define FCH  128
#define LN_EPS 1e-5f

typedef __attribute__((ext_vector_type(16))) __bf16 v16bf;
typedef __attribute__((ext_vector_type(8)))  float  v8f;

__device__ __forceinline__ unsigned short f2bf(float f) {
  union { float f; unsigned u; } v; v.f = f;
  unsigned r = v.u + 0x7FFFu + ((v.u >> 16) & 1u);  // round-to-nearest-even
  return (unsigned short)(r >> 16);
}

__device__ __forceinline__ float gelu_exact(float x) {
  return 0.5f * x * (1.0f + erff(x * 0.70710678118654752440f));
}

// LDS byte offset = low 32 bits of the generic pointer (LDS aperture keeps offset in addr[31:0]).
__device__ __forceinline__ uint32_t lds_lo32(const void* p) {
  return (uint32_t)(uintptr_t)p;
}

// Per-lane 32B async global->LDS copy (GVS form: SGPR base + u32 VGPR offset).
// The 24-bit inst offset is added to BOTH the LDS and global address (ISA 08 §4.4).
__device__ __forceinline__ void stage32B(uint32_t lds, uint32_t voff, const unsigned short* sbase) {
  asm volatile(
      "global_load_async_to_lds_b128 %0, %1, %2\n\t"
      "global_load_async_to_lds_b128 %0, %1, %2 offset:16"
      :: "v"(lds), "v"(voff), "s"(sbase) : "memory");
}
__device__ __forceinline__ void wait_async0() {
  asm volatile("s_wait_asynccnt 0" ::: "memory");
}
__device__ __forceinline__ void wait_async2() {  // allow the prefetched panel to stay in flight
  asm volatile("s_wait_asynccnt 2" ::: "memory");
}

// ---------------------------------------------------------------- init (zero y + counts)
__global__ void moe_init_kernel(float* p, int n) {
  int i = blockIdx.x * blockDim.x + threadIdx.x;
  if (i < n) p[i] = 0.0f;
}

// ---------------------------------------------------------------- x fp32 -> bf16
__global__ __launch_bounds__(256) void moe_cvt_x_kernel(const float* __restrict__ in,
                                                        unsigned short* __restrict__ out) {
  int i = (blockIdx.x * 256 + threadIdx.x) * 4;
  float4 v = *(const float4*)(in + i);
  out[i + 0] = f2bf(v.x);
  out[i + 1] = f2bf(v.y);
  out[i + 2] = f2bf(v.z);
  out[i + 3] = f2bf(v.w);
}

// ------------------------------------------------- weights: fp32 [R][C] -> bf16 [C][R]
__global__ __launch_bounds__(256) void moe_cvt_transpose_kernel(
    const float* __restrict__ in, unsigned short* __restrict__ out, int R, int C) {
  __shared__ float tile[32][33];
  const int ez = blockIdx.z;
  const float*     src = in  + (size_t)ez * R * C;
  unsigned short*  dst = out + (size_t)ez * R * C;
  const int r0 = blockIdx.y * 32, c0 = blockIdx.x * 32;
  for (int i = threadIdx.x; i < 1024; i += 256) {
    int r = i >> 5, c = i & 31;
    tile[r][c] = src[(size_t)(r0 + r) * C + c0 + c];
  }
  __syncthreads();
  for (int i = threadIdx.x; i < 1024; i += 256) {
    int r = i >> 5, c = i & 31;
    dst[(size_t)(c0 + r) * R + r0 + c] = f2bf(tile[c][r]);
  }
}

// ---------------------------------------------------------------- router (1 wave / token)
__global__ __launch_bounds__(32) void moe_router_kernel(
    const float* __restrict__ x, const float* __restrict__ Wr,
    float* __restrict__ probs_all, int* __restrict__ counts,
    int* __restrict__ lists, float* __restrict__ wgts) {
  const int t = blockIdx.x;
  const int lane = threadIdx.x;
  const float* xr = x + (size_t)t * DDIM;
  float p[NEXP];
#pragma unroll
  for (int e = 0; e < NEXP; ++e) p[e] = 0.0f;
  for (int i = 0; i < DDIM / 32; ++i) {
    float xv = xr[i * 32 + lane];
#pragma unroll
    for (int e = 0; e < NEXP; ++e) p[e] += xv * Wr[e * DDIM + i * 32 + lane];
  }
#pragma unroll
  for (int e = 0; e < NEXP; ++e) {
#pragma unroll
    for (int off = 16; off > 0; off >>= 1) p[e] += __shfl_xor(p[e], off, 32);
  }
  if (lane == 0) {
    float mx = p[0];
#pragma unroll
    for (int e = 1; e < NEXP; ++e) mx = fmaxf(mx, p[e]);
    float pr[NEXP], s = 0.0f;
#pragma unroll
    for (int e = 0; e < NEXP; ++e) { pr[e] = expf(p[e] - mx); s += pr[e]; }
    float inv = 1.0f / s;
#pragma unroll
    for (int e = 0; e < NEXP; ++e) {
      pr[e] *= inv;
      probs_all[(size_t)t * NEXP + e] = pr[e];
    }
    int i1 = 0;
#pragma unroll
    for (int e = 1; e < NEXP; ++e) if (pr[e] > pr[i1]) i1 = e;   // first-index tie-break
    int i2 = (i1 == 0) ? 1 : 0;
#pragma unroll
    for (int e = 0; e < NEXP; ++e) if (e != i1 && pr[e] > pr[i2]) i2 = e;
    float dn = pr[i1] + pr[i2];
    float w1v = pr[i1] / dn, w2v = pr[i2] / dn;
    int p1 = atomicAdd(&counts[i1], 1);
    lists[i1 * NTOK + p1] = t; wgts[i1 * NTOK + p1] = w1v;
    int p2 = atomicAdd(&counts[i2], 1);
    lists[i2 * NTOK + p2] = t; wgts[i2 * NTOK + p2] = w2v;
  }
}

// ---------------- grouped FFN GEMMs: per-wave double-buffered async panels, no panel barriers
__global__ __launch_bounds__(256) void moe_ffn_kernel(
    const unsigned short* __restrict__ xb,   // [T][D] bf16
    const unsigned short* __restrict__ W1t,  // [E][F][D] bf16 (row f, K contiguous)
    const float* __restrict__ b1,
    const unsigned short* __restrict__ W2t,  // [E][D][F] bf16 (row d, K contiguous)
    const float* __restrict__ b2,
    const int* __restrict__ counts, const int* __restrict__ lists,
    const float* __restrict__ wgts, float* __restrict__ y) {
  const int e   = blockIdx.y;
  const int cnt = counts[e];
  const int m0  = blockIdx.x * 16;
  if (m0 >= cnt) return;

  __shared__ __align__(32) unsigned short xA[16 * DDIM];     // 32 KB  shared bf16 A-tile
  __shared__ __align__(32) unsigned short wB[8 * 2 * 512];   // 16 KB  [wave][buf][16r x 32k]
  __shared__ __align__(32) unsigned short hA[16 * FCH];      //  4 KB  shared hidden tile
  __shared__ int   stok[16];
  __shared__ float swgt[16];

  const int tid  = threadIdx.x;
  const int wv   = tid >> 5;     // wave 0..7 (wave32)
  const int lane = tid & 31;
  const int lN   = lane & 15;
  const int lH   = lane >> 4;

  if (tid < 16) {
    int m = tid;
    if (m0 + m < cnt) {
      stok[m] = lists[e * NTOK + m0 + m];
      swgt[m] = wgts[e * NTOK + m0 + m];
    } else { stok[m] = 0; swgt[m] = 0.0f; }
  }
  __syncthreads();

  // gather 16 bf16 token rows into LDS (8 x 16B per thread, one address, inst offsets)
  {
    int m  = tid >> 4;
    int c0 = (tid & 15) * 64;                      // ushort elements within the row
    const unsigned short* src = xb + (size_t)stok[m] * DDIM + c0;
    uint32_t ldsx = lds_lo32(&xA[m * DDIM + c0]);
    asm volatile(
        "global_load_async_to_lds_b128 %0, %1, off\n\t"
        "global_load_async_to_lds_b128 %0, %1, off offset:16\n\t"
        "global_load_async_to_lds_b128 %0, %1, off offset:32\n\t"
        "global_load_async_to_lds_b128 %0, %1, off offset:48\n\t"
        "global_load_async_to_lds_b128 %0, %1, off offset:64\n\t"
        "global_load_async_to_lds_b128 %0, %1, off offset:80\n\t"
        "global_load_async_to_lds_b128 %0, %1, off offset:96\n\t"
        "global_load_async_to_lds_b128 %0, %1, off offset:112"
        :: "v"(ldsx), "v"(src) : "memory");
  }
  wait_async0();
  __syncthreads();

  // thread-invariant staging offsets: lane covers 32B (2 chunks) of its wave's 1 KB panel
  const uint32_t vo1 = (uint32_t)(((wv * 16 + (lane >> 1)) * DDIM + (lane & 1) * 16) * 2);
  const uint32_t vo2 = (uint32_t)(((wv * 16 + (lane >> 1)) * FDIM + (lane & 1) * 16) * 2);
  uint32_t ldsw[2];
  ldsw[0] = lds_lo32(&wB[(wv * 2 + 0) * 512]) + (lane >> 1) * 64 + (lane & 1) * 32;
  ldsw[1] = ldsw[0] + 1024;

  const v8f vzero = {0.f, 0.f, 0.f, 0.f, 0.f, 0.f, 0.f, 0.f};
  v8f acc2[8];
#pragma unroll
  for (int i = 0; i < 8; ++i) acc2[i] = vzero;

#pragma unroll 1
  for (int fc = 0; fc < FDIM; fc += FCH) {
    // ---- GEMM1: h(16 x 128) = xA(16 x 1024) * W1[e][:, fc:fc+128]; 32 K-panels, depth-2 pipe
    const unsigned short* w1base = W1t + ((size_t)e * FDIM + fc) * DDIM;
    v8f acc1a = vzero, acc1b = vzero;
    stage32B(ldsw[0], vo1, w1base);
    stage32B(ldsw[1], vo1, w1base + 32);
#pragma unroll
    for (int p = 0; p < 32; ++p) {
      if (p < 31) wait_async2(); else wait_async0();
      v16bf a = *(const v16bf*)&xA[lN * DDIM + p * 32 + lH * 16];
      v16bf b = *(const v16bf*)&wB[(wv * 2 + (p & 1)) * 512 + lN * 32 + lH * 16];
      if (p & 1)
        acc1b = __builtin_amdgcn_wmma_f32_16x16x32_bf16(false, a, false, b,
                                                        (short)0, acc1b, false, false);
      else
        acc1a = __builtin_amdgcn_wmma_f32_16x16x32_bf16(false, a, false, b,
                                                        (short)0, acc1a, false, false);
      if (p + 2 < 32) stage32B(ldsw[p & 1], vo1, w1base + (p + 2) * 32);
    }

    // prefetch first two GEMM2 panels (overlaps the GELU below). p = kk*8 + tt (K-major)
    const unsigned short* w2base = W2t + (size_t)e * DDIM * FDIM + fc;
    stage32B(ldsw[0], vo2, w2base);                          // p=0: kk=0, tt=0
    stage32B(ldsw[1], vo2, w2base + (size_t)128 * FDIM);     // p=1: kk=0, tt=1

    // bias + exact GELU, publish bf16 hidden tile
    float b1v = b1[(size_t)e * FDIM + fc + wv * 16 + lN];
    __syncthreads();                                         // prev chunk's hA reads complete
#pragma unroll
    for (int j = 0; j < 8; ++j) {
      float hv = gelu_exact(acc1a[j] + acc1b[j] + b1v);
      hA[(j + lH * 8) * FCH + wv * 16 + lN] = f2bf(hv);
    }
    __syncthreads();                                         // publish hA to all waves

    // preload the 4 A-operands (K-blocks of hA) into registers once
    v16bf a2[4];
#pragma unroll
    for (int kk = 0; kk < 4; ++kk)
      a2[kk] = *(const v16bf*)&hA[lN * FCH + kk * 32 + lH * 16];

    // ---- GEMM2: acc2 += hA(16 x 128) * W2[e][fc:fc+128, :]; 32 panels, 8 indep acc chains
#pragma unroll
    for (int kk = 0; kk < 4; ++kk) {
#pragma unroll
      for (int tt = 0; tt < 8; ++tt) {
        int p = kk * 8 + tt;
        if (p < 31) wait_async2(); else wait_async0();
        v16bf b = *(const v16bf*)&wB[(wv * 2 + (p & 1)) * 512 + lN * 32 + lH * 16];
        acc2[tt] = __builtin_amdgcn_wmma_f32_16x16x32_bf16(false, a2[kk], false, b,
                                                           (short)0, acc2[tt], false, false);
        int pn = p + 2;
        if (pn < 32)
          stage32B(ldsw[p & 1], vo2,
                   w2base + (size_t)(pn & 7) * 128 * FDIM + (pn >> 3) * 32);
      }
    }
  }

  // epilogue: + b2, scale by combine weight, scatter-add (exactly 2 commutative adds/element)
#pragma unroll
  for (int tt = 0; tt < 8; ++tt) {
    int d = tt * 128 + wv * 16 + lN;
    float b2v = b2[(size_t)e * DDIM + d];
#pragma unroll
    for (int j = 0; j < 8; ++j) {
      int m = j + lH * 8;
      if (m0 + m < cnt)
        atomicAdd(&y[(size_t)stok[m] * DDIM + d], (acc2[tt][j] + b2v) * swgt[m]);
    }
  }
}

// ---------------------------------------------------------------- layernorm
__global__ __launch_bounds__(256) void moe_ln_kernel(const float* __restrict__ y,
                                                     const float* __restrict__ lnw,
                                                     const float* __restrict__ lnb,
                                                     float* __restrict__ out) {
  const int t = blockIdx.x;
  const int tid = threadIdx.x;
  const int lane = tid & 31, warp = tid >> 5;
  const float* yr = y + (size_t)t * DDIM;
  float v[4], s = 0.f, s2 = 0.f;
#pragma unroll
  for (int i = 0; i < 4; ++i) {
    v[i] = yr[tid + i * 256];
    s += v[i];
    s2 += v[i] * v[i];
  }
#pragma unroll
  for (int off = 16; off > 0; off >>= 1) {
    s  += __shfl_xor(s,  off, 32);
    s2 += __shfl_xor(s2, off, 32);
  }
  __shared__ float rs[8], rs2[8];
  if (lane == 0) { rs[warp] = s; rs2[warp] = s2; }
  __syncthreads();
  float ts = 0.f, ts2 = 0.f;
#pragma unroll
  for (int w = 0; w < 8; ++w) { ts += rs[w]; ts2 += rs2[w]; }
  float mu  = ts * (1.0f / DDIM);
  float var = ts2 * (1.0f / DDIM) - mu * mu;
  float inv = rsqrtf(var + LN_EPS);
#pragma unroll
  for (int i = 0; i < 4; ++i) {
    int dIdx = tid + i * 256;
    out[(size_t)t * DDIM + dIdx] = (v[i] - mu) * inv * lnw[dIdx] + lnb[dIdx];
  }
}

// ---------------------------------------------------------------- aux loss (deterministic reduce)
__global__ __launch_bounds__(256) void moe_aux_kernel(const float* __restrict__ probs_all,
                                                      const int* __restrict__ counts,
                                                      float* __restrict__ out_aux) {
  __shared__ float red[256];
  __shared__ float sp[NEXP];
  const int tid = threadIdx.x;
  for (int e = 0; e < NEXP; ++e) {
    float s = 0.0f;
    for (int t = tid; t < NTOK; t += 256) s += probs_all[(size_t)t * NEXP + e];
    red[tid] = s;
    __syncthreads();
    for (int off = 128; off > 0; off >>= 1) {
      if (tid < off) red[tid] += red[tid + off];
      __syncthreads();
    }
    if (tid == 0) sp[e] = red[0];
    __syncthreads();
  }
  if (tid == 0) {
    float bal = 0.0f, imp2 = 0.0f;
    for (int e = 0; e < NEXP; ++e) {
      float density = sp[e] / (float)NTOK;
      float usage   = (float)counts[e] / (float)NTOK;
      bal  += density * usage;
      imp2 += sp[e] * sp[e];
    }
    out_aux[0] = bal * (float)NEXP + imp2 / (float)NEXP;
  }
}

// ---------------------------------------------------------------- launch
extern "C" void kernel_launch(void* const* d_in, const int* in_sizes, int n_in,
                              void* d_out, int out_size, void* d_ws, size_t ws_size,
                              hipStream_t stream) {
  const float* x   = (const float*)d_in[0];
  const float* Wr  = (const float*)d_in[1];
  const float* W1  = (const float*)d_in[2];
  const float* b1  = (const float*)d_in[3];
  const float* W2  = (const float*)d_in[4];
  const float* b2  = (const float*)d_in[5];
  const float* lnw = (const float*)d_in[6];
  const float* lnb = (const float*)d_in[7];
  float* out = (float*)d_out;

  // ws layout: y[T*D] f32 | counts[8] | lists[8T] | wgts[8T] | probs_all[8T]
  //            | xb[T*D] bf16 | W1t[E*F*D] bf16 | W2t[E*D*F] bf16   (~153 MB)
  float* y         = (float*)d_ws;
  int*   counts    = (int*)(y + (size_t)NTOK * DDIM);
  int*   lists     = counts + NEXP;
  float* wgts      = (float*)(lists + NEXP * NTOK);
  float* probs_all = wgts + NEXP * NTOK;
  unsigned short* xb  = (unsigned short*)(probs_all + (size_t)NTOK * NEXP);
  unsigned short* W1t = xb  + (size_t)NTOK * DDIM;
  unsigned short* W2t = W1t + (size_t)NEXP * FDIM * DDIM;

  const int zeroN = NTOK * DDIM + NEXP;
  moe_init_kernel<<<(zeroN + 255) / 256, 256, 0, stream>>>((float*)d_ws, zeroN);
  moe_cvt_x_kernel<<<(NTOK * DDIM) / 1024, 256, 0, stream>>>(x, xb);
  // W1[e]: [D][F] -> W1t[e]: [F][D];  W2[e]: [F][D] -> W2t[e]: [D][F]
  moe_cvt_transpose_kernel<<<dim3(FDIM / 32, DDIM / 32, NEXP), 256, 0, stream>>>(W1, W1t, DDIM, FDIM);
  moe_cvt_transpose_kernel<<<dim3(DDIM / 32, FDIM / 32, NEXP), 256, 0, stream>>>(W2, W2t, FDIM, DDIM);
  moe_router_kernel<<<NTOK, 32, 0, stream>>>(x, Wr, probs_all, counts, lists, wgts);
  moe_ffn_kernel<<<dim3(NTOK / 16, NEXP), 256, 0, stream>>>(xb, W1t, b1, W2t, b2,
                                                            counts, lists, wgts, y);
  moe_ln_kernel<<<NTOK, 256, 0, stream>>>(y, lnw, lnb, out);
  moe_aux_kernel<<<1, 256, 0, stream>>>(probs_all, counts, out + (size_t)NTOK * DDIM);
}
